// LSHLinear_61529701483101
// MI455X (gfx1250) — compile-verified, matrix operand order checked by reading, not run on hardware
//
#include <hip/hip_runtime.h>

typedef __attribute__((ext_vector_type(16))) __bf16 v16bf;
typedef __attribute__((ext_vector_type(8)))  __bf16 v8bf;
typedef __attribute__((ext_vector_type(4)))  __bf16 v4bf;
typedef __attribute__((ext_vector_type(8)))  float  v8f;

#define D_DIM 1024
#define N_DIM 32768
#define M_DIM 2048
#define TS    40   // padded LDS row stride (bf16 elems): 80B -> 16B aligned, conflict-free

// ---------------------------------------------------------------------------
// SimHash codes: one thread per row of V [nrows, 1024]; R is [64, 1024].
// Output: per-row uint64, byte t = bucket id of table t (exact fp32 signs).
// ---------------------------------------------------------------------------
__global__ void __launch_bounds__(256)
codes_kernel(const float* __restrict__ V, const float* __restrict__ R,
             unsigned long long* __restrict__ codes) {
  __shared__ float Vs[256 * 33];   // 33 = pad vs bank conflicts on per-thread reads
  __shared__ float Rs[64 * 32];    // broadcast reads, no pad needed
  const int tid = threadIdx.x;
  const int rowBase = blockIdx.x * 256;

  float acc[64];
#pragma unroll
  for (int j = 0; j < 64; ++j) acc[j] = 0.0f;

  for (int d0 = 0; d0 < D_DIM; d0 += 32) {
    __syncthreads();
#pragma unroll
    for (int i = 0; i < 32; ++i) {               // 256x32 V tile, coalesced over d
      int flat = tid + i * 256;
      int r = flat >> 5, dd = flat & 31;
      Vs[r * 33 + dd] = V[(size_t)(rowBase + r) * D_DIM + d0 + dd];
    }
#pragma unroll
    for (int i = 0; i < 8; ++i) {                // 64x32 R tile
      int flat = tid + i * 256;
      int j = flat >> 5, dd = flat & 31;
      Rs[j * 32 + dd] = R[(size_t)j * D_DIM + d0 + dd];
    }
    __syncthreads();
    for (int dd = 0; dd < 32; ++dd) {
      float w = Vs[tid * 33 + dd];
#pragma unroll
      for (int j = 0; j < 64; ++j) acc[j] = fmaf(w, Rs[j * 32 + dd], acc[j]);
    }
  }

  unsigned long long code = 0ULL;
#pragma unroll
  for (int t = 0; t < 8; ++t) {
    unsigned c = 0;
#pragma unroll
    for (int h = 0; h < 8; ++h)
      if (acc[t * 8 + h] > 0.0f) c |= (1u << h);
    code |= (unsigned long long)c << (8 * t);
  }
  codes[rowBase + tid] = code;
}

// ---------------------------------------------------------------------------
// GEMM helpers
// ---------------------------------------------------------------------------
union Frag { v16bf v; v8bf h[2]; };

// Load one 16x32 bf16 fragment from LDS per the documented CDNA5 16-bit A/B
// layout: lane L holds row L%16; lanes 0-15 get K=0..7 & 16..23,
// lanes 16-31 get K=8..15 & 24..31. Two 16B ds loads per lane.
__device__ __forceinline__ v16bf ldfrag(const __bf16* base, int lane) {
  const __bf16* p = base + (lane & 15) * TS + ((lane >> 4) << 3);
  Frag f;
  f.h[0] = *(const v8bf*)(p);
  f.h[1] = *(const v8bf*)(p + 16);
  return f.v;
}

// Stage a 128x32 fp32 tile from global into bf16 hi/lo LDS tiles.
__device__ __forceinline__ void stage_tile(const float* __restrict__ g,
                                           int rowBase, int k0,
                                           __bf16* s_hi, __bf16* s_lo, int tid) {
#pragma unroll
  for (int i = 0; i < 4; ++i) {
    int flat = tid + i * 256;          // float4 index, 0..1023
    int r  = flat >> 3;                // 0..127
    int kc = (flat & 7) << 2;          // 0,4,...,28
    const float4 v = *(const float4*)(g + (size_t)(rowBase + r) * D_DIM + k0 + kc);
    __bf16 h0 = (__bf16)v.x, h1 = (__bf16)v.y, h2 = (__bf16)v.z, h3 = (__bf16)v.w;
    v4bf hv = {h0, h1, h2, h3};
    v4bf lv = {(__bf16)(v.x - (float)h0), (__bf16)(v.y - (float)h1),
               (__bf16)(v.z - (float)h2), (__bf16)(v.w - (float)h3)};
    *(v4bf*)(s_hi + r * TS + kc) = hv;   // 8B ds stores, aligned
    *(v4bf*)(s_lo + r * TS + kc) = lv;
  }
}

// ---------------------------------------------------------------------------
// Masked GEMM: out[r,c] = anyByteMatch(icodes[r], wcodes[c]) ? X.Wt + b : 0
// 128x128 block tile, 8 wave32s, wave tile 64x32 (4x2 fragments of 16x16),
// split-bf16 (hi/lo) accumulation: hh + hl + lh per fragment per K-step.
// ---------------------------------------------------------------------------
__global__ void __launch_bounds__(256)
lsh_gemm(const float* __restrict__ X, const float* __restrict__ Wt,
         const float* __restrict__ bias,
         const unsigned long long* __restrict__ icodes,
         const unsigned long long* __restrict__ wcodes,
         float* __restrict__ out) {
  __shared__ __bf16 sAh[128 * TS];
  __shared__ __bf16 sAl[128 * TS];
  __shared__ __bf16 sBh[128 * TS];
  __shared__ __bf16 sBl[128 * TS];

  const int tid  = threadIdx.x;
  const int lane = tid & 31;
  const int wave = tid >> 5;
  const int wr   = wave >> 2;       // 0..1 : wave row
  const int wcn  = wave & 3;        // 0..3 : wave col
  const int mBase = blockIdx.y * 128;
  const int nBase = blockIdx.x * 128;

  v8f acc[4][2];
#pragma unroll
  for (int m = 0; m < 4; ++m)
#pragma unroll
    for (int n = 0; n < 2; ++n)
#pragma unroll
      for (int j = 0; j < 8; ++j) acc[m][n][j] = 0.0f;

  for (int k0 = 0; k0 < D_DIM; k0 += 32) {
    __syncthreads();
    stage_tile(X,  mBase, k0, sAh, sAl, tid);
    stage_tile(Wt, nBase, k0, sBh, sBl, tid);
    __syncthreads();

    v16bf bh[2], bl[2];
#pragma unroll
    for (int n = 0; n < 2; ++n) {
      bh[n] = ldfrag(sBh + (wcn * 32 + n * 16) * TS, lane);
      bl[n] = ldfrag(sBl + (wcn * 32 + n * 16) * TS, lane);
    }
#pragma unroll
    for (int m = 0; m < 4; ++m) {
      v16bf ah = ldfrag(sAh + (wr * 64 + m * 16) * TS, lane);
      v16bf al = ldfrag(sAl + (wr * 64 + m * 16) * TS, lane);
#pragma unroll
      for (int n = 0; n < 2; ++n) {
        acc[m][n] = __builtin_amdgcn_wmma_f32_16x16x32_bf16(
            false, ah, false, bh[n], (short)0, acc[m][n], false, false);
        acc[m][n] = __builtin_amdgcn_wmma_f32_16x16x32_bf16(
            false, ah, false, bl[n], (short)0, acc[m][n], false, false);
        acc[m][n] = __builtin_amdgcn_wmma_f32_16x16x32_bf16(
            false, al, false, bh[n], (short)0, acc[m][n], false, false);
      }
    }
  }

  // Epilogue: bias + LSH bucket-match mask + store.
  // C layout: VGPR j -> lanes 0-15: (M=j, N=lane), lanes 16-31: (M=j+8, N=lane-16)
  const int lrow  = lane & 15;
  const int lhalf = (lane >> 4) << 3;
#pragma unroll
  for (int m = 0; m < 4; ++m) {
#pragma unroll
    for (int n = 0; n < 2; ++n) {
      const int col = nBase + wcn * 32 + n * 16 + lrow;
      const unsigned long long wc64 = wcodes[col];
      const float bv = bias[col];
#pragma unroll
      for (int j = 0; j < 8; ++j) {
        const int row = mBase + wr * 64 + m * 16 + lhalf + j;
        const unsigned long long x64 = icodes[row] ^ wc64;
        // "any byte is zero" => some table's buckets match
        const bool match =
            (((x64 - 0x0101010101010101ULL) & ~x64) & 0x8080808080808080ULL) != 0ULL;
        out[(size_t)row * N_DIM + col] = match ? (acc[m][n][j] + bv) : 0.0f;
      }
    }
  }
}

// ---------------------------------------------------------------------------
extern "C" void kernel_launch(void* const* d_in, const int* in_sizes, int n_in,
                              void* d_out, int out_size, void* d_ws, size_t ws_size,
                              hipStream_t stream) {
  const float* x = (const float*)d_in[0];   // [4,512,1024] -> [2048,1024]
  const float* W = (const float*)d_in[1];   // [32768,1024]
  const float* b = (const float*)d_in[2];   // [32768]
  const float* R = (const float*)d_in[3];   // [8,8,1024] -> [64,1024]
  float* out = (float*)d_out;               // [2048,32768]

  unsigned long long* wcodes = (unsigned long long*)d_ws;          // 32768 * 8B
  unsigned long long* icodes = wcodes + N_DIM;                     // 2048  * 8B

  codes_kernel<<<N_DIM / 256, 256, 0, stream>>>(W, R, wcodes);
  codes_kernel<<<M_DIM / 256, 256, 0, stream>>>(x, R, icodes);

  dim3 grid(N_DIM / 128, M_DIM / 128);      // (256, 16)
  lsh_gemm<<<grid, 256, 0, stream>>>(x, W, b, icodes, wcodes, out);
}
